// EQGATLocalGNN_55783035240748
// MI455X (gfx1250) — matrix operand order repaired and checked by reading, not compiled, not run on hardware
//
#include <hip/hip_runtime.h>

// ---------------------------------------------------------------------------
// EQGAT-style local GNN for MI455X (gfx1250).
// Heavy path: fused per-edge MLP, per block = 256 edges:
//   - weights TDM-loaded to LDS once per block (tensor_load_to_lds, TENSORcnt)
//   - edge features gathered via GLOBAL_LOAD_ASYNC_TO_LDS_B128 (ASYNCcnt)
//   - GEMMs via v_wmma_f32_16x16x32_bf16, f32 accumulate
//   - atomic scatter into f32 accumulators
// ---------------------------------------------------------------------------

#define N_NODES 50000
#define N_EDGES 800000
#define N_GRAPH 64
#define SI 64
#define VI 16
#define EDGE_DIM 16
#define KP1 160        // 146 padded to multiple of 32
#define W2R 96         // GEMM2 output rows padded (80 or 96 -> 96)
#define W1_ELEMS (SI * KP1)   // 10240 bf16
#define W2_ELEMS (W2R * SI)   // 6144 bf16
#define EPB 256        // edges per block (16 tiles of 16)

typedef __attribute__((ext_vector_type(16))) __bf16 v16bf;
typedef __attribute__((ext_vector_type(8)))  __bf16 v8bf;
typedef __attribute__((ext_vector_type(8)))  float  v8f;
typedef unsigned int u32x4 __attribute__((ext_vector_type(4)));
typedef int          i32x8 __attribute__((ext_vector_type(8)));
typedef int          i32x4 __attribute__((ext_vector_type(4)));
typedef __attribute__((address_space(3))) void lds_void;

__device__ __forceinline__ float siluf(float x) { return x / (1.f + __expf(-x)); }

__device__ __forceinline__ unsigned lds_offset_of(const void* p) {
    return (unsigned)(__UINTPTR_TYPE__)(lds_void*)p;
}

// 1-D bf16 tile copy global -> LDS through the Tensor Data Mover.
// D# layout per CDNA5 ISA 8.3/8.4: count=1, type=2 (image), data_size=2B,
// tensor_dim0 = tile_dim0 = nelem, tensor_dim1 = tile_dim1 = 1.
// 6-arg builtin form (clang-23 / therock headers).
__device__ __forceinline__ void tdm_load_1d_bf16(const void* gptr, unsigned lds_addr,
                                                 unsigned nelem) {
    unsigned long long ga = (unsigned long long)gptr;
    u32x4 g0;
    g0[0] = 1u;                                             // count=1, user mode
    g0[1] = lds_addr;                                       // lds_addr
    g0[2] = (unsigned)(ga & 0xffffffffu);                   // global_addr lo
    g0[3] = (unsigned)((ga >> 32) & 0x01ffffffu) | (2u << 30); // addr hi | type=2
    i32x8 g1;
    g1[0] = (int)(1u << 16);                                // data_size=1 (2 bytes)
    g1[1] = (int)((nelem & 0xffffu) << 16);                 // tensor_dim0[15:0]
    g1[2] = (int)((nelem >> 16) & 0xffffu) | (1 << 16);     // dim0 hi | tensor_dim1=1
    g1[3] = (int)((nelem & 0xffffu) << 16);                 // dim1 hi=0 | tile_dim0
    g1[4] = 1;                                              // tile_dim1=1, tile_dim2=0
    g1[5] = (int)nelem;                                     // tensor_dim0_stride lo
    g1[6] = 0;
    g1[7] = 0;
    i32x4 z4 = {0, 0, 0, 0};
    i32x8 z8 = {0, 0, 0, 0, 0, 0, 0, 0};
    __builtin_amdgcn_tensor_load_to_lds(g0, g1, z4, z4, z8, 0);
}

// ---------------------------------------------------------------------------
// utility kernels
// ---------------------------------------------------------------------------
__global__ void k_fill(float* p, int n) {
    int i = blockIdx.x * blockDim.x + threadIdx.x;
    if (i < n) p[i] = 0.f;
}

__global__ void k_deg(const int* __restrict__ dst, float* __restrict__ deg, int ne) {
    int i = blockIdx.x * blockDim.x + threadIdx.x;
    if (i < ne) atomicAdd(&deg[dst[i]], 1.f);
}

// W1: [k1][n1] -> W1t: [n1][KP1] ; W2: [k2][n2] -> W2t: [W2R][k2]   (bf16)
__global__ void k_wconv(const float* __restrict__ W1, const float* __restrict__ W2,
                        __bf16* __restrict__ W1t, __bf16* __restrict__ W2t,
                        int k1, int n1, int k2, int n2) {
    int idx = blockIdx.x * blockDim.x + threadIdx.x;
    int tot1 = n1 * KP1;
    if (idx < tot1) {
        int o = idx / KP1, k = idx % KP1;
        W1t[idx] = (k < k1) ? (__bf16)W1[(size_t)k * n1 + o] : (__bf16)0.f;
    } else {
        idx -= tot1;
        if (idx >= W2R * k2) return;
        int o = idx / k2, k = idx % k2;
        W2t[idx] = (o < n2) ? (__bf16)W2[(size_t)k * n2 + o] : (__bf16)0.f;
    }
}

// ---------------------------------------------------------------------------
// layer norm (per-graph stats via atomics; G=64)
// ---------------------------------------------------------------------------
__global__ void k_ln_sum(const float* __restrict__ s, const int* __restrict__ batch,
                         float* __restrict__ gsum, float* __restrict__ gcnt, int n) {
    int i = blockIdx.x * blockDim.x + threadIdx.x;
    if (i >= n) return;
    const float* row = s + (size_t)i * SI;
    float acc = 0.f;
    for (int j = 0; j < SI; ++j) acc += row[j];
    int g = batch[i];
    atomicAdd(&gsum[g], acc);
    atomicAdd(&gcnt[g], 1.f);
}

__global__ void k_ln_var(const float* __restrict__ s, const float* __restrict__ v,
                         const int* __restrict__ batch,
                         const float* __restrict__ gsum, const float* __restrict__ gcnt,
                         float* __restrict__ gvar, float* __restrict__ gv, int n) {
    int i = blockIdx.x * blockDim.x + threadIdx.x;
    if (i >= n) return;
    int g = batch[i];
    float cnt  = fmaxf(gcnt[g], 1.f);
    float mean = gsum[g] / (cnt * (float)SI);
    const float* row = s + (size_t)i * SI;
    float ssq = 0.f;
    for (int j = 0; j < SI; ++j) { float dd = row[j] - mean; ssq += dd * dd; }
    atomicAdd(&gvar[g], ssq * (1.f / (float)SI));
    const float* vr = v + (size_t)i * 3 * VI;
    float vn = 0.f;
    for (int k = 0; k < 3 * VI; ++k) { float t = vr[k]; vn += t * t; }
    atomicAdd(&gv[g], vn * (1.f / (float)VI));
}

__global__ void k_ln_apply(const float* __restrict__ s_src, const float* __restrict__ v_src,
                           const int* __restrict__ batch,
                           const float* __restrict__ w, const float* __restrict__ b,
                           const float* __restrict__ gsum, const float* __restrict__ gcnt,
                           const float* __restrict__ gvar, const float* __restrict__ gv,
                           float* __restrict__ s_dst, float* __restrict__ v_dst,
                           __bf16* __restrict__ sbf, int n) {
    int idx = blockIdx.x * blockDim.x + threadIdx.x;
    int total = n * (SI + 3 * VI);
    if (idx >= total) return;
    int node = idx / (SI + 3 * VI);
    int j    = idx % (SI + 3 * VI);
    int g = batch[node];
    float cnt = fmaxf(gcnt[g], 1.f);
    if (j < SI) {
        float mean = gsum[g] / (cnt * (float)SI);
        float var  = fmaxf(gvar[g] / cnt, 1e-6f);
        float val  = (s_src[(size_t)node * SI + j] - mean) * rsqrtf(var) * w[j] + b[j];
        s_dst[(size_t)node * SI + j] = val;
        if (sbf) sbf[(size_t)node * SI + j] = (__bf16)val;
    } else {
        int k = j - SI;
        float vm = fmaxf(gv[g] / cnt, 1e-6f);
        v_dst[(size_t)node * 48 + k] = v_src[(size_t)node * 48 + k] * rsqrtf(vm);
    }
}

// ---------------------------------------------------------------------------
// node pre-projections:  sb = s @ Wsc + bsc ;  vb = v @ Wv
// ---------------------------------------------------------------------------
__global__ void k_sb(const float* __restrict__ s, const float* __restrict__ Wsc,
                     const float* __restrict__ bsc, float* __restrict__ sb, int n) {
    int idx = blockIdx.x * blockDim.x + threadIdx.x;
    if (idx >= n * SI) return;
    int node = idx / SI, o = idx % SI;
    const float* row = s + (size_t)node * SI;
    float acc = bsc[o];
    for (int k = 0; k < SI; ++k) acc += row[k] * Wsc[(size_t)k * SI + o];
    sb[idx] = acc;
}

__global__ void k_vb(const float* __restrict__ v, const float* __restrict__ Wv,
                     float* __restrict__ vb, int n) {
    int idx = blockIdx.x * blockDim.x + threadIdx.x;
    if (idx >= n * 48) return;
    int node = idx / 48, rem = idx % 48;
    int c = rem / VI, i = rem % VI;
    const float* row = v + ((size_t)node * 3 + c) * VI;
    float acc = 0.f;
    for (int j = 0; j < VI; ++j) acc += row[j] * Wv[(size_t)j * VI + i];
    vb[idx] = acc;
}

// ---------------------------------------------------------------------------
// fused WMMA edge-message kernel
//   block = 128 threads (4 waves), 256 edges = 16 tiles of 16 edges.
//   Weights TDM-loaded into LDS once; edge features async-gathered into LDS.
// ---------------------------------------------------------------------------
__global__ __launch_bounds__(128)
void k_edge(const __bf16* __restrict__ sbf, const float* __restrict__ sb,
            const float* __restrict__ vb,
            const int* __restrict__ src, const int* __restrict__ dst,
            const float* __restrict__ d, const float* __restrict__ a,
            const float* __restrict__ r, const float* __restrict__ ef,
            const __bf16* __restrict__ W1t, const float* __restrict__ b1,
            const __bf16* __restrict__ W2t, const float* __restrict__ b2,
            float* __restrict__ s_acc, float* __restrict__ v_acc,
            int ne, int msg_width, int has_v_in) {
    __shared__ __bf16 w1s[SI][KP1];   // 20480 B
    __shared__ __bf16 w2s[W2R][SI];   // 12288 B
    __shared__ __bf16 xs[16][KP1];    //  5120 B
    __shared__ __bf16 hs[16][SI];     //  2048 B
    __shared__ float  ms[16][W2R];    //  6144 B

    const int tid  = threadIdx.x;
    const int wave = tid >> 5, lane = tid & 31;
    const int half = lane >> 4, ln = lane & 15;

    // ---- TDM: pull both weight matrices into LDS (wave 0 issues) ----
    if (tid < 32) {
        tdm_load_1d_bf16(W1t, lds_offset_of(&w1s[0][0]), W1_ELEMS);
        tdm_load_1d_bf16(W2t, lds_offset_of(&w2s[0][0]), W2_ELEMS);
        __builtin_amdgcn_s_wait_tensorcnt(0);
    }
    __syncthreads();

    const unsigned xs_base = lds_offset_of(&xs[0][0]);

    for (int tile = 0; tile < EPB / 16; ++tile) {
        const int e0 = blockIdx.x * EPB + tile * 16;

        // ---- async gather: s[dst] | s[src] rows (128 bf16 cols) -> LDS ----
        // 16 edges x 2 halves x 8 chunks of 16B = 256 B128 transfers.
        for (int idx = tid; idx < 256; idx += 128) {
            int ed = idx >> 4, chunk = idx & 15;
            int ge = e0 + ed;
            if (ge >= ne) ge = ne - 1;
            int hsel = chunk >> 3, sub = chunk & 7;
            int node = hsel ? src[ge] : dst[ge];
            const __bf16* gp = sbf + (size_t)node * SI + sub * 8;
            unsigned lo = xs_base + (unsigned)(ed * (KP1 * 2) + hsel * 128 + sub * 16);
            asm volatile("global_load_async_to_lds_b128 %0, %1, off"
                         :: "v"(lo), "v"(gp) : "memory");
        }
        // ---- tail columns 128..159: d, a, e, pad (converted to bf16) ----
        for (int idx = tid; idx < 16 * 32; idx += 128) {
            int ed = idx >> 5, col = 128 + (idx & 31);
            int ge = e0 + ed;
            __bf16 bv = (__bf16)0.f;
            if (ge < ne) {
                if (col == 128)      bv = (__bf16)d[ge];
                else if (col == 129) bv = (__bf16)a[ge];
                else if (col < 146)  bv = (__bf16)ef[(size_t)ge * EDGE_DIM + (col - 130)];
            }
            xs[ed][col] = bv;
        }
        asm volatile("s_wait_asynccnt 0" ::: "memory");
        __syncthreads();

        // ---- GEMM1: each wave owns one 16-col tile of h ----
        {
            const int nt = wave;
            v8f acc;
            float bias = b1[nt * 16 + ln];
            #pragma unroll
            for (int i = 0; i < 8; ++i) acc[i] = bias;

            #pragma unroll
            for (int kb = 0; kb < KP1; kb += 32) {
                v16bf A, B;
                {
                    const __bf16* row = &xs[ln][0];
                    v8bf lo = *(const v8bf*)(row + kb + half * 8);
                    v8bf hi = *(const v8bf*)(row + kb + 16 + half * 8);
                    #pragma unroll
                    for (int i = 0; i < 8; ++i) { A[i] = lo[i]; A[8 + i] = hi[i]; }
                }
                {
                    const __bf16* wr = &w1s[nt * 16 + ln][kb + half * 8];
                    v8bf lo = *(const v8bf*)(wr);
                    v8bf hi = *(const v8bf*)(wr + 16);
                    #pragma unroll
                    for (int i = 0; i < 8; ++i) { B[i] = lo[i]; B[8 + i] = hi[i]; }
                }
                acc = __builtin_amdgcn_wmma_f32_16x16x32_bf16(
                    false, A, false, B, (short)0, acc, false, false);
            }
            #pragma unroll
            for (int i = 0; i < 8; ++i)
                hs[i + half * 8][nt * 16 + ln] = (__bf16)siluf(acc[i]);
        }
        __syncthreads();

        // ---- GEMM2: 6 tiles of 16 cols over 4 waves ----
        for (int nt = wave; nt < 6; nt += 4) {
            v8f acc;
            float bias = (nt * 16 + ln < msg_width) ? b2[nt * 16 + ln] : 0.f;
            #pragma unroll
            for (int i = 0; i < 8; ++i) acc[i] = bias;

            #pragma unroll
            for (int kb = 0; kb < SI; kb += 32) {
                v16bf A, B;
                {
                    const __bf16* row = &hs[ln][0];
                    v8bf lo = *(const v8bf*)(row + kb + half * 8);
                    v8bf hi = *(const v8bf*)(row + kb + 16 + half * 8);
                    #pragma unroll
                    for (int i = 0; i < 8; ++i) { A[i] = lo[i]; A[8 + i] = hi[i]; }
                }
                {
                    const __bf16* wr = &w2s[nt * 16 + ln][kb + half * 8];
                    v8bf lo = *(const v8bf*)(wr);
                    v8bf hi = *(const v8bf*)(wr + 16);
                    #pragma unroll
                    for (int i = 0; i < 8; ++i) { B[i] = lo[i]; B[8 + i] = hi[i]; }
                }
                acc = __builtin_amdgcn_wmma_f32_16x16x32_bf16(
                    false, A, false, B, (short)0, acc, false, false);
            }
            #pragma unroll
            for (int i = 0; i < 8; ++i) ms[i + half * 8][nt * 16 + ln] = acc[i];
        }
        __syncthreads();

        // ---- scatter s:  s_acc[dst] += aij * sb[src] ----
        for (int idx = tid; idx < 16 * SI; idx += 128) {
            int ed = idx >> 6, col = idx & 63;
            int ge = e0 + ed;
            if (ge < ne) {
                int sn = src[ge], dn = dst[ge];
                float ns = ms[ed][col] * sb[(size_t)sn * SI + col];
                atomicAdd(&s_acc[(size_t)dn * SI + col], ns);
            }
        }
        // ---- scatter v:  v_acc[dst] += r*vij_a (+ vij_b * vb[src]) ----
        for (int idx = tid; idx < 16 * VI; idx += 128) {
            int ed = idx >> 4, i = idx & 15;
            int ge = e0 + ed;
            if (ge < ne) {
                int sn = src[ge], dn = dst[ge];
                float va  = ms[ed][SI + i];
                float vbm = has_v_in ? ms[ed][SI + VI + i] : 0.f;
                #pragma unroll
                for (int c = 0; c < 3; ++c) {
                    float nv = r[(size_t)ge * 3 + c] * va;
                    if (has_v_in) nv += vbm * vb[((size_t)sn * 3 + c) * VI + i];
                    atomicAdd(&v_acc[((size_t)dn * 3 + c) * VI + i], nv);
                }
            }
        }
        __syncthreads();   // protect xs/hs/ms before next tile
    }
}

// ---------------------------------------------------------------------------
// post-aggregation + gated equivariant block
// ---------------------------------------------------------------------------
__global__ void k_finalize(float* __restrict__ s, float* __restrict__ v,
                           const float* __restrict__ s_acc, const float* __restrict__ v_acc,
                           const float* __restrict__ deg, int n) {
    int idx = blockIdx.x * blockDim.x + threadIdx.x;
    if (idx >= n * (SI + 48)) return;
    int node = idx / (SI + 48), j = idx % (SI + 48);
    if (j < SI) {
        s[(size_t)node * SI + j] += s_acc[(size_t)node * SI + j];
    } else {
        int k = j - SI;
        float inv = 1.f / fmaxf(deg[node], 1.f);
        v[(size_t)node * 48 + k] += v_acc[(size_t)node * 48 + k] * inv;
    }
}

__global__ void k_vv(const float* __restrict__ v, const float* __restrict__ Wv0,
                     float* __restrict__ vv, int n) {
    int idx = blockIdx.x * blockDim.x + threadIdx.x;
    if (idx >= n * 96) return;
    int node = idx / 96, rem = idx % 96;
    int c = rem / 32, o = rem % 32;
    const float* row = v + ((size_t)node * 3 + c) * VI;
    float acc = 0.f;
    for (int k = 0; k < VI; ++k) acc += row[k] * Wv0[(size_t)k * 32 + o];
    vv[idx] = acc;
}

__global__ void k_xcat(const float* __restrict__ s, const float* __restrict__ vv,
                       float* __restrict__ x, int n) {
    int idx = blockIdx.x * blockDim.x + threadIdx.x;
    if (idx >= n * 80) return;
    int node = idx / 80, j = idx % 80;
    if (j < SI) {
        x[idx] = s[(size_t)node * SI + j];
    } else {
        int i = j - SI;
        float ss = 0.f;
        #pragma unroll
        for (int c = 0; c < 3; ++c) {
            float t = vv[(size_t)node * 96 + c * 32 + i];
            ss += t * t;
        }
        x[idx] = sqrtf(fmaxf(ss, 1e-6f));
    }
}

__global__ void k_hmlp(const float* __restrict__ x, const float* __restrict__ W,
                       const float* __restrict__ b, float* __restrict__ h, int n) {
    int idx = blockIdx.x * blockDim.x + threadIdx.x;
    if (idx >= n * SI) return;
    int node = idx / SI, o = idx % SI;
    const float* row = x + (size_t)node * 80;
    float acc = b[o];
    for (int k = 0; k < 80; ++k) acc += row[k] * W[(size_t)k * SI + o];
    h[idx] = siluf(acc);
}

__global__ void k_outp(const float* __restrict__ in, const float* __restrict__ W,
                       const float* __restrict__ b, float* __restrict__ out,
                       int n, int kdim) {
    int idx = blockIdx.x * blockDim.x + threadIdx.x;
    if (idx >= n * 80) return;
    int node = idx / 80, j = idx % 80;
    const float* row = in + (size_t)node * kdim;
    float acc = b[j];
    for (int k = 0; k < kdim; ++k) acc += row[k] * W[(size_t)k * 80 + j];
    out[idx] = acc;
}

__global__ void k_update(float* __restrict__ s, float* __restrict__ v,
                         const float* __restrict__ out, const float* __restrict__ vv,
                         const float* __restrict__ Wv1, int n, int use_mlp) {
    int idx = blockIdx.x * blockDim.x + threadIdx.x;
    if (idx >= n * (SI + 48)) return;
    int node = idx / (SI + 48), j = idx % (SI + 48);
    if (j < SI) {
        s[(size_t)node * SI + j] += out[(size_t)node * 80 + VI + j];
    } else {
        int rem = j - SI;
        int c = rem / VI, o = rem % VI;
        float dv;
        if (use_mlp) {
            dv = 0.f;
            for (int i = 0; i < VI; ++i) {
                float g  = out[(size_t)node * 80 + i];
                float vd = vv[(size_t)node * 96 + c * 32 + VI + i];
                dv += g * vd * Wv1[(size_t)i * VI + o];
            }
        } else {
            dv = out[(size_t)node * 80 + o] * vv[(size_t)node * 96 + c * 32 + VI + o];
        }
        v[((size_t)node * 3 + c) * VI + o] += dv;
    }
}

// ---------------------------------------------------------------------------
// host orchestration
// ---------------------------------------------------------------------------
struct ConvP {
    const float *We1, *be1, *We2, *be2, *Wsc, *bsc, *Wv0;
    const float *Wv;                    // layers > 0
    const float *Ws1, *bs1, *Ws2, *bs2, *Wv1;  // mlp layers
    const float *Ws, *bs;               // last layer
};

extern "C" void kernel_launch(void* const* d_in, const int* in_sizes, int n_in,
                              void* d_out, int out_size, void* d_ws, size_t ws_size,
                              hipStream_t stream) {
    const int N = N_NODES, E = N_EDGES, G = N_GRAPH;
    const float* s_in  = (const float*)d_in[0];
    const float* v_in  = (const float*)d_in[1];
    const int*   eidx  = (const int*)d_in[3];
    const int*   batch = (const int*)d_in[4];
    const float* d_e   = (const float*)d_in[5];
    const float* a_e   = (const float*)d_in[6];
    const float* r_e   = (const float*)d_in[7];
    const float* e_e   = (const float*)d_in[8];
    const int* src = eidx;
    const int* dst = eidx + E;

    // ---- parse params (pytree / sorted-key flattening) ----
    int p = 9;
    auto F = [&](void) { return (const float*)d_in[p++]; };
    ConvP conv[5]; const float *nw[5], *nb[5], *ow, *ob;
    for (int l = 0; l < 5; ++l) {
        bool hv = l > 0, mlp = l < 4;
        ConvP& c = conv[l];
        c.We1 = F(); c.We2 = F();
        if (mlp) { c.Ws1 = F(); c.Ws2 = F(); } else { c.Ws = F(); }
        c.Wsc = F();
        if (hv) c.Wv = F();
        c.Wv0 = F();
        if (mlp) c.Wv1 = F();
        c.be1 = F(); c.be2 = F();
        if (mlp) { c.bs1 = F(); c.bs2 = F(); } else { c.bs = F(); }
        c.bsc = F();
    }
    for (int l = 0; l < 5; ++l) { nb[l] = F(); nw[l] = F(); }
    ob = F(); ow = F();

    // ---- workspace layout ----
    char* base = (char*)d_ws;
    size_t off = 0;
    auto alloc = [&](size_t bytes) -> void* {
        off = (off + 255) & ~(size_t)255;
        void* q = base + off;
        off += bytes;
        return q;
    };
    float* deg    = (float*)alloc((size_t)N * 4);
    float* stats  = (float*)alloc((size_t)4 * G * 4);   // sum | cnt | var | vsum
    float* gsum = stats, *gcnt = stats + G, *gvar = stats + 2 * G, *gv = stats + 3 * G;
    float* s_cur  = (float*)alloc((size_t)N * SI * 4);
    float* v_cur  = (float*)alloc((size_t)N * 48 * 4);
    __bf16* s_bf  = (__bf16*)alloc((size_t)N * SI * 2);
    float* sb     = (float*)alloc((size_t)N * SI * 4);
    float* vb     = (float*)alloc((size_t)N * 48 * 4);
    float* s_acc  = (float*)alloc((size_t)N * SI * 4);
    float* v_acc  = (float*)alloc((size_t)N * 48 * 4);
    float* vv     = (float*)alloc((size_t)N * 96 * 4);
    float* xbuf   = (float*)alloc((size_t)N * 80 * 4);
    float* hbuf   = (float*)alloc((size_t)N * SI * 4);
    float* obuf   = (float*)alloc((size_t)N * 80 * 4);
    __bf16* W1t[5]; __bf16* W2t[5];
    for (int l = 0; l < 5; ++l) {
        W1t[l] = (__bf16*)alloc((size_t)W1_ELEMS * 2);
        W2t[l] = (__bf16*)alloc((size_t)W2_ELEMS * 2);
    }

    auto blocks = [](int total, int bs) { return (total + bs - 1) / bs; };

    // ---- degrees (constant across layers) ----
    k_fill<<<blocks(N, 256), 256, 0, stream>>>(deg, N);
    k_deg<<<blocks(E, 256), 256, 0, stream>>>(dst, deg, E);

    // ---- weight conversion (bf16, transposed, padded) ----
    for (int l = 0; l < 5; ++l) {
        int width = (l == 0) ? 80 : 96;
        int tot = W1_ELEMS + W2_ELEMS;
        k_wconv<<<blocks(tot, 256), 256, 0, stream>>>(
            conv[l].We1, conv[l].We2, W1t[l], W2t[l],
            2 * SI + EDGE_DIM + 2, SI, SI, width);
    }

    // ---- layers ----
    for (int l = 0; l < 5; ++l) {
        bool hv = l > 0, mlp = l < 4;
        int width = (l == 0) ? 80 : 96;
        const float* ssrc = (l == 0) ? s_in : s_cur;
        const float* vsrc = (l == 0) ? v_in : v_cur;

        // layer norm
        k_fill<<<1, 256, 0, stream>>>(stats, 4 * G);
        k_ln_sum<<<blocks(N, 256), 256, 0, stream>>>(ssrc, batch, gsum, gcnt, N);
        k_ln_var<<<blocks(N, 256), 256, 0, stream>>>(ssrc, vsrc, batch, gsum, gcnt, gvar, gv, N);
        k_ln_apply<<<blocks(N * (SI + 48), 256), 256, 0, stream>>>(
            ssrc, vsrc, batch, nw[l], nb[l], gsum, gcnt, gvar, gv,
            s_cur, v_cur, s_bf, N);

        // node pre-projections
        k_sb<<<blocks(N * SI, 256), 256, 0, stream>>>(s_cur, conv[l].Wsc, conv[l].bsc, sb, N);
        const float* vb_use = v_cur;
        if (hv) {
            k_vb<<<blocks(N * 48, 256), 256, 0, stream>>>(v_cur, conv[l].Wv, vb, N);
            vb_use = vb;
        }

        // edge messages (TDM + async + WMMA) + scatter
        k_fill<<<blocks(N * SI, 256), 256, 0, stream>>>(s_acc, N * SI);
        k_fill<<<blocks(N * 48, 256), 256, 0, stream>>>(v_acc, N * 48);
        k_edge<<<blocks(E, EPB), 128, 0, stream>>>(
            s_bf, sb, vb_use, src, dst, d_e, a_e, r_e, e_e,
            W1t[l], conv[l].be1, W2t[l], conv[l].be2,
            s_acc, v_acc, E, width, hv ? 1 : 0);
        k_finalize<<<blocks(N * (SI + 48), 256), 256, 0, stream>>>(
            s_cur, v_cur, s_acc, v_acc, deg, N);

        // gated equivariant update
        k_vv<<<blocks(N * 96, 256), 256, 0, stream>>>(v_cur, conv[l].Wv0, vv, N);
        k_xcat<<<blocks(N * 80, 256), 256, 0, stream>>>(s_cur, vv, xbuf, N);
        if (mlp) {
            k_hmlp<<<blocks(N * SI, 256), 256, 0, stream>>>(xbuf, conv[l].Ws1, conv[l].bs1, hbuf, N);
            k_outp<<<blocks(N * 80, 256), 256, 0, stream>>>(hbuf, conv[l].Ws2, conv[l].bs2, obuf, N, SI);
        } else {
            k_outp<<<blocks(N * 80, 256), 256, 0, stream>>>(xbuf, conv[l].Ws, conv[l].bs, obuf, N, 80);
        }
        k_update<<<blocks(N * (SI + 48), 256), 256, 0, stream>>>(
            s_cur, v_cur, obuf, vv, mlp ? conv[l].Wv1 : (const float*)nullptr, N, mlp ? 1 : 0);
    }

    // ---- out_norm -> d_out (s then v, flat) ----
    float* s_out = (float*)d_out;
    float* v_out = (float*)d_out + (size_t)N * SI;
    k_fill<<<1, 256, 0, stream>>>(stats, 4 * G);
    k_ln_sum<<<blocks(N, 256), 256, 0, stream>>>(s_cur, batch, gsum, gcnt, N);
    k_ln_var<<<blocks(N, 256), 256, 0, stream>>>(s_cur, v_cur, batch, gsum, gcnt, gvar, gv, N);
    k_ln_apply<<<blocks(N * (SI + 48), 256), 256, 0, stream>>>(
        s_cur, v_cur, batch, ow, ob, gsum, gcnt, gvar, gv,
        s_out, v_out, (__bf16*)nullptr, N);
}
